// NeighborhoodAttention2DEfficient_21388937134560
// MI455X (gfx1250) — matrix operand order repaired and checked by reading, not compiled, not run on hardware
//
#include <hip/hip_runtime.h>
#include <hip/hip_bf16.h>

typedef __attribute__((ext_vector_type(16))) __bf16 v16bf;
typedef __attribute__((ext_vector_type(8)))  __bf16 bf16x8;
typedef __attribute__((ext_vector_type(4)))  __bf16 bf16x4;
typedef __attribute__((ext_vector_type(8)))  float  v8f;

#define BM 128
#define BN 128
#define BK 32
#define LDP 40   // padded row length in bf16 (80 B = 5*16 B: keeps b128 alignment, spreads banks)

__device__ __forceinline__ float f4get(const float4& v, int c) {
    return c == 0 ? v.x : c == 1 ? v.y : c == 2 ? v.z : v.w;
}

// C[M,N] = A[M,K] * B[K,N] + bias[N]; A/B fp32 row-major; bf16 WMMA, fp32 accum.
// Requires M % 128 == 0, N % 128 == 0, K % 32 == 0.
__global__ __launch_bounds__(256)
void gemm_bias_bf16(const float* __restrict__ A, const float* __restrict__ Bm,
                    const float* __restrict__ bias, float* __restrict__ C,
                    int M, int N, int K) {
    // lA: row-major [m][k] ; lB: TRANSPOSED [n][k] so B-fragments are contiguous per lane
    __shared__ __bf16 lA[2][BM][LDP];
    __shared__ __bf16 lB[2][BN][LDP];

    const int t     = threadIdx.x;
    const int lane  = t & 31;
    const int wid   = t >> 5;       // 8 waves
    const int wm    = wid >> 1;     // 0..3 : 32-row band
    const int wn    = wid & 1;      // 0..1 : 64-col band
    const int row0  = blockIdx.y * BM;
    const int col0  = blockIdx.x * BN;

    const int lrow  = lane & 15;    // row (A) / col (B,C) within 16
    const int lhalf = lane >> 4;    // K-half (A,B) / M-half (C)

    v8f acc[2][4];
    #pragma unroll
    for (int mt = 0; mt < 2; ++mt)
        #pragma unroll
        for (int nt = 0; nt < 4; ++nt)
            #pragma unroll
            for (int r = 0; r < 8; ++r)
                acc[mt][nt][r] = 0.0f;

    const int nK = K / BK;
    float4 ar[4];           // A: 4 strips of (row, kquad)
    float4 br[4];           // B: one 4k x 4c block (row j = k0+bkq*4+j, cols bcq*4..+3)
    const int bkq = t >> 5;     // 0..7  : k-quad of this thread's B block
    const int bcq = t & 31;     // 0..31 : c-quad of this thread's B block

    // ---- vectorized global loads of one K-tile into registers ----
    auto loadTile = [&](int k0) {
        #pragma unroll
        for (int s = 0; s < 4; ++s) {               // A tile 128x32: thread -> (row, kquad)
            int idx = s * 256 + t;
            int r = idx >> 3, kq = idx & 7;
            ar[s] = *(const float4*)(A + (size_t)(row0 + r) * K + k0 + kq * 4);
        }
        #pragma unroll
        for (int j = 0; j < 4; ++j)                 // B tile 32x128: 4 coalesced rows of the 4x4 block
            br[j] = *(const float4*)(Bm + (size_t)(k0 + bkq * 4 + j) * N + col0 + bcq * 4);
    };
    // ---- convert + store registers into LDS buffer ----
    auto storeTile = [&](int buf) {
        #pragma unroll
        for (int s = 0; s < 4; ++s) {
            int idx = s * 256 + t;
            int r = idx >> 3, kq = idx & 7;
            bf16x4 v = { (__bf16)ar[s].x, (__bf16)ar[s].y, (__bf16)ar[s].z, (__bf16)ar[s].w };
            *(bf16x4*)(&lA[buf][r][kq * 4]) = v;            // ds_store_b64
        }
        #pragma unroll
        for (int c = 0; c < 4; ++c) {                       // transpose 4x4 block in registers
            bf16x4 v = { (__bf16)f4get(br[0], c), (__bf16)f4get(br[1], c),
                         (__bf16)f4get(br[2], c), (__bf16)f4get(br[3], c) };
            *(bf16x4*)(&lB[buf][bcq * 4 + c][bkq * 4]) = v; // ds_store_b64 (k-contiguous)
        }
    };

    loadTile(0);
    storeTile(0);
    int buf = 0;

    for (int kt = 0; kt < nK; ++kt) {
        const bool more = (kt + 1) < nK;
        if (more) loadTile((kt + 1) * BK);          // global loads in flight during compute
        __syncthreads();                             // LDS buf ready for all waves

        #pragma unroll
        for (int mt = 0; mt < 2; ++mt) {
            // A fragment 16x32: lanes 0-15 (row=lane): K 0..7 & 16..23 ; lanes 16-31: K 8..15 & 24..31
            const int arow = wm * 32 + mt * 16 + lrow;
            bf16x8 a_lo = *(const bf16x8*)(&lA[buf][arow][lhalf * 8]);        // ds_load_b128
            bf16x8 a_hi = *(const bf16x8*)(&lA[buf][arow][16 + lhalf * 8]);   // ds_load_b128
            v16bf af = __builtin_shufflevector(a_lo, a_hi,
                0, 1, 2, 3, 4, 5, 6, 7, 8, 9, 10, 11, 12, 13, 14, 15);
            #pragma unroll
            for (int nt = 0; nt < 4; ++nt) {
                // B fragment 32x16: col = lane&15, lanes 0-15: K 0..7 & 16..23 ; lanes 16-31: K 8..15 & 24..31
                const int bcol = wn * 64 + nt * 16 + lrow;
                bf16x8 b_lo = *(const bf16x8*)(&lB[buf][bcol][lhalf * 8]);      // ds_load_b128
                bf16x8 b_hi = *(const bf16x8*)(&lB[buf][bcol][16 + lhalf * 8]);  // ds_load_b128
                v16bf bfr = __builtin_shufflevector(b_lo, b_hi,
                    0, 1, 2, 3, 4, 5, 6, 7, 8, 9, 10, 11, 12, 13, 14, 15);
                acc[mt][nt] = __builtin_amdgcn_wmma_f32_16x16x32_bf16(
                    false, af, false, bfr, (short)0, acc[mt][nt], false, false);
            }
        }

        if (more) storeTile(buf ^ 1);               // fill other buffer; safe: all waves passed sync
        buf ^= 1;
    }

    // ---- epilogue: C layout lanes 0-15 N=lane M=r, lanes 16-31 N=lane-16 M=r+8 ----
    #pragma unroll
    for (int mt = 0; mt < 2; ++mt) {
        #pragma unroll
        for (int nt = 0; nt < 4; ++nt) {
            const int col = col0 + wn * 64 + nt * 16 + lrow;
            const float bv = bias[col];
            const int rbase = row0 + wm * 32 + mt * 16 + lhalf * 8;
            #pragma unroll
            for (int r = 0; r < 8; ++r)
                C[(size_t)(rbase + r) * N + col] = acc[mt][nt][r] + bv;
        }
    }
}

// Neighborhood attention 7x7, one thread per (pixel, head).
// qkv layout per pixel: [q(256) | k(256) | v(256)], head_dim 32.
// Online softmax; zero-padded OOB (logit = rpb only, v = 0) to match reference.
__global__ __launch_bounds__(256)
void natten7x7(const float* __restrict__ qkv, const float* __restrict__ rpb,
               float* __restrict__ out) {
    const int tid = blockIdx.x * 256 + threadIdx.x;
    if (tid >= 4 * 56 * 56 * 8) return;

    const int head = tid & 7;
    const int pix  = tid >> 3;          // 0..12543
    const int w    = pix % 56;
    const int h    = (pix / 56) % 56;
    const int b    = pix / (56 * 56);

    const float scale = 0.17677669529663687f;   // 32^-0.5
    const float* qp = qkv + (size_t)pix * 768 + head * 32;

    float q[32];
    #pragma unroll
    for (int d = 0; d < 32; ++d) q[d] = qp[d] * scale;

    float m = -3.0e38f, l = 0.0f;
    float acc[32];
    #pragma unroll
    for (int d = 0; d < 32; ++d) acc[d] = 0.0f;

    const float* rp = rpb + head * 49;

    for (int ki = 0; ki < 7; ++ki) {
        const int hh = h - 3 + ki;
        for (int kj = 0; kj < 7; ++kj) {
            const int ww = w - 3 + kj;
            const bool inb = (hh >= 0) & (hh < 56) & (ww >= 0) & (ww < 56);
            float s = rp[ki * 7 + kj];
            const float* kp = qkv;  // dummy init
            if (inb) {
                const int npix = (b * 56 + hh) * 56 + ww;
                kp = qkv + (size_t)npix * 768 + 256 + head * 32;
                float dot = 0.0f;
                #pragma unroll
                for (int d = 0; d < 32; ++d) dot += q[d] * kp[d];
                s += dot;
            }
            const float mnew = fmaxf(m, s);
            const float corr = __expf(m - mnew);
            const float p    = __expf(s - mnew);
            l = l * corr + p;
            m = mnew;
            if (inb) {
                const float* vp = kp + 256;
                #pragma unroll
                for (int d = 0; d < 32; ++d) acc[d] = acc[d] * corr + p * vp[d];
            } else {
                #pragma unroll
                for (int d = 0; d < 32; ++d) acc[d] = acc[d] * corr;
            }
        }
    }

    const float inv = 1.0f / l;
    float* op = out + (size_t)pix * 256 + head * 32;
    #pragma unroll
    for (int d = 0; d < 32; ++d) op[d] = acc[d] * inv;
}

extern "C" void kernel_launch(void* const* d_in, const int* in_sizes, int n_in,
                              void* d_out, int out_size, void* d_ws, size_t ws_size,
                              hipStream_t stream) {
    (void)in_sizes; (void)n_in; (void)out_size; (void)ws_size;

    const float* x      = (const float*)d_in[0];   // [12544, 256]
    const float* w_qkv  = (const float*)d_in[1];   // [256, 768]
    const float* b_qkv  = (const float*)d_in[2];   // [768]
    const float* rpb    = (const float*)d_in[3];   // [8, 49]
    const float* w_proj = (const float*)d_in[4];   // [256, 256]
    const float* b_proj = (const float*)d_in[5];   // [256]
    float* out = (float*)d_out;                    // [12544, 256]

    const int M = 4 * 56 * 56;                     // 12544 = 98 * 128
    float* qkv      = (float*)d_ws;                // 12544*768 fp32
    float* attn_out = qkv + (size_t)M * 768;       // 12544*256 fp32

    // 1) QKV projection: [M,256] x [256,768] + bias
    dim3 g1(768 / BN, M / BM);
    gemm_bias_bf16<<<g1, 256, 0, stream>>>(x, w_qkv, b_qkv, qkv, M, 768, 256);

    // 2) Neighborhood attention
    const int nthreads = M * 8;
    natten7x7<<<(nthreads + 255) / 256, 256, 0, stream>>>(qkv, rpb, attn_out);

    // 3) Output projection: [M,256] x [256,256] + bias
    dim3 g2(256 / BN, M / BM);
    gemm_bias_bf16<<<g2, 256, 0, stream>>>(attn_out, w_proj, b_proj, out, M, 256, 256);
}
